// MECP_GAP_Model_51299089384085
// MI455X (gfx1250) — compile-verified, matrix operand order checked by reading, not run on hardware
//
#include <hip/hip_runtime.h>
#include <hip/hip_bf16.h>
#include <math.h>

// Problem constants (match reference)
#define NN 50000          // nodes
#define EE 800000         // edges
#define NH 4              // heads
#define HD 32             // per-head dim
#define FF 128            // hidden feats = NH*HD = IN
#define PP 8              // partitions
#define NEG_SLOPE 0.2f

typedef _Float16 v8h  __attribute__((ext_vector_type(8)));
typedef _Float16 v16h __attribute__((ext_vector_type(16)));
typedef float    v8f  __attribute__((ext_vector_type(8)));

// ---- order-preserving float<->uint mapping (for atomic segment-max) ----
__device__ __forceinline__ unsigned f2ord(float f) {
    unsigned u = __float_as_uint(f);
    return (u & 0x80000000u) ? ~u : (u | 0x80000000u);
}
__device__ __forceinline__ float ord2f(unsigned u) {
    return __uint_as_float((u & 0x80000000u) ? (u & 0x7fffffffu) : ~u);
}

// ---- convert node features (N,128) f32 -> f16 ----
__global__ void k_cvt_x(const float* __restrict__ x, _Float16* __restrict__ xh, int n) {
    int t = blockIdx.x * blockDim.x + threadIdx.x;
    if (t < n) xh[t] = (_Float16)x[t];
}

// ---- convert W (H,128,32) f32 -> WT f16 with WT[c][k], c = h*32+d ----
__global__ void k_cvt_w(const float* __restrict__ W, _Float16* __restrict__ WT) {
    int t = blockIdx.x * blockDim.x + threadIdx.x;
    if (t >= FF * FF) return;
    int c = t >> 7, k = t & 127;
    WT[t] = (_Float16)W[(c >> 5) * (FF * HD) + k * HD + (c & 31)];
}

// ---- WMMA GEMM: Wh(N,128) = Xh(N,128) * WT^T ; one wave per 16x16 tile ----
// A 16-bit layout: lane m=L&15, K-offset = (L>=16)*8; element j -> K = (j<8?j:j+8)+off
// B assumed mirrored with N = L&15 (WT stored contiguous in K).
__global__ void __launch_bounds__(256) k_gemm(const _Float16* __restrict__ Xh,
                                              const _Float16* __restrict__ WT,
                                              float* __restrict__ Wh) {
    const int wave = threadIdx.x >> 5;
    const int lane = threadIdx.x & 31;
    const int r = lane & 15;
    const int hs = lane >> 4;          // 0 or 1 -> K offset +8
    const int mbase = blockIdx.x * 16;
    const int nbase = wave * 16;

    const _Float16* arow = Xh + (size_t)(mbase + r) * FF;
    const _Float16* brow = WT + (size_t)(nbase + r) * FF;

    v8f c = {};
#pragma unroll
    for (int kk = 0; kk < FF; kk += 32) {
        v8h alo = *(const v8h*)(arow + kk + hs * 8);
        v8h ahi = *(const v8h*)(arow + kk + 16 + hs * 8);
        v8h blo = *(const v8h*)(brow + kk + hs * 8);
        v8h bhi = *(const v8h*)(brow + kk + 16 + hs * 8);
        v16h a, b;
#pragma unroll
        for (int i = 0; i < 8; ++i) {
            a[i] = alo[i]; a[i + 8] = ahi[i];
            b[i] = blo[i]; b[i + 8] = bhi[i];
        }
        c = __builtin_amdgcn_wmma_f32_16x16x32_f16(false, a, false, b,
                                                   (short)0, c, false, false);
    }
    // C/D layout: VGPR i -> M = i + hs*8 ; N = lane&15
#pragma unroll
    for (int i = 0; i < 8; ++i)
        Wh[(size_t)(mbase + i + hs * 8) * FF + nbase + r] = c[i];
}

// ---- per-(node,head) attention dots e_l, e_r ----
__global__ void k_elr(const float* __restrict__ Wh, const float* __restrict__ al,
                      const float* __restrict__ ar, float* __restrict__ el,
                      float* __restrict__ er) {
    int t = blockIdx.x * blockDim.x + threadIdx.x;
    if (t >= NN * NH) return;
    int n = t >> 2, h = t & 3;
    const float* w = Wh + (size_t)n * FF + h * HD;
    float sl = 0.f, sr = 0.f;
#pragma unroll
    for (int d = 0; d < HD; ++d) {
        sl += w[d] * al[h * HD + d];
        sr += w[d] * ar[h * HD + d];
    }
    el[t] = sl;
    er[t] = sr;
}

// ---- per-layer accumulator init ----
__global__ void k_init(unsigned* __restrict__ menc, float* __restrict__ den,
                       float* __restrict__ acc) {
    int t = blockIdx.x * blockDim.x + threadIdx.x;
    if (t >= NN * FF) return;
    acc[t] = 0.f;
    if (t < NN * NH) {
        menc[t] = f2ord(-INFINITY);
        den[t] = 0.f;
    }
}

// ---- log(clip(edge_weight)) once ----
__global__ void k_logw(const float* __restrict__ ew, float* __restrict__ lw) {
    int t = blockIdx.x * blockDim.x + threadIdx.x;
    if (t < EE) lw[t] = logf(fmaxf(ew[t], 1e-8f));
}

// ---- edge pass 1: score + segment max ----
__global__ void k_edge1(const int* __restrict__ row, const int* __restrict__ col,
                        const float* __restrict__ el, const float* __restrict__ er,
                        const float* __restrict__ lw, float* __restrict__ S,
                        unsigned* __restrict__ menc) {
    int t = blockIdx.x * blockDim.x + threadIdx.x;
    if (t >= EE * NH) return;
    int e = t >> 2, h = t & 3;
    int r = row[e], c = col[e];
    float s = el[r * NH + h] + er[c * NH + h];
    s = (s >= 0.f) ? s : NEG_SLOPE * s;
    s += lw[e];
    S[t] = s;
    atomicMax(&menc[c * NH + h], f2ord(s));
}

// ---- edge pass 2: exp + segment sum ----
__global__ void k_edge2(const int* __restrict__ col, float* __restrict__ S,
                        const unsigned* __restrict__ menc, float* __restrict__ den) {
    int t = blockIdx.x * blockDim.x + threadIdx.x;
    if (t >= EE * NH) return;
    int e = t >> 2, h = t & 3;
    int c = col[e];
    float a = __expf(S[t] - ord2f(menc[c * NH + h]));
    S[t] = a;
    atomicAdd(&den[c * NH + h], a);
}

// ---- edge pass 3: weighted message scatter-add, one wave per edge ----
__global__ void k_edge3(const int* __restrict__ row, const int* __restrict__ col,
                        const float* __restrict__ S, const float* __restrict__ den,
                        const float* __restrict__ Wh, float* __restrict__ acc) {
    int t = blockIdx.x * blockDim.x + threadIdx.x;
    if (t >= EE * 32) return;
    int e = t >> 5;
    int lane = t & 31;
    int h = lane >> 3;
    int q = (lane & 7) << 2;           // 4 dims per lane
    int r = row[e], c = col[e];
    float coef = S[e * NH + h] / (den[c * NH + h] + 1e-8f);
    float4 w = *(const float4*)(Wh + (size_t)r * FF + h * HD + q);
    float* dst = acc + (size_t)c * FF + h * HD + q;
    atomicAdd(dst + 0, coef * w.x);
    atomicAdd(dst + 1, coef * w.y);
    atomicAdd(dst + 2, coef * w.z);
    atomicAdd(dst + 3, coef * w.w);
}

// ---- layer-1 finalize: +bias, relu, -> f16 input for layer 2 ----
__global__ void k_fin1(const float* __restrict__ acc, const float* __restrict__ b,
                       _Float16* __restrict__ xh) {
    int t = blockIdx.x * blockDim.x + threadIdx.x;
    if (t >= NN * FF) return;
    float v = acc[t] + b[t & 127];
    xh[t] = (_Float16)fmaxf(v, 0.f);
}

// ---- layer-2 finalize fused with projection + softmax ----
__global__ void k_proj(const float* __restrict__ acc, const float* __restrict__ b,
                       const float* __restrict__ Wp, const float* __restrict__ bp,
                       float* __restrict__ out) {
    int n = blockIdx.x * blockDim.x + threadIdx.x;
    if (n >= NN) return;
    float lg[PP];
#pragma unroll
    for (int p = 0; p < PP; ++p) lg[p] = bp[p];
    const float* arow = acc + (size_t)n * FF;
    for (int c = 0; c < FF; ++c) {
        float v = arow[c] + b[c];
#pragma unroll
        for (int p = 0; p < PP; ++p) lg[p] += v * Wp[c * PP + p];
    }
    float mx = lg[0];
#pragma unroll
    for (int p = 1; p < PP; ++p) mx = fmaxf(mx, lg[p]);
    float sum = 0.f;
#pragma unroll
    for (int p = 0; p < PP; ++p) {
        lg[p] = __expf(lg[p] - mx);
        sum += lg[p];
    }
    float inv = 1.f / sum;
#pragma unroll
    for (int p = 0; p < PP; ++p) out[n * PP + p] = lg[p] * inv;
}

extern "C" void kernel_launch(void* const* d_in, const int* in_sizes, int n_in,
                              void* d_out, int out_size, void* d_ws, size_t ws_size,
                              hipStream_t stream) {
    const float* x   = (const float*)d_in[0];
    const int*   ei  = (const int*)d_in[1];
    const float* ew  = (const float*)d_in[2];
    const float* W1  = (const float*)d_in[3];
    const float* al1 = (const float*)d_in[4];
    const float* ar1 = (const float*)d_in[5];
    const float* b1  = (const float*)d_in[6];
    const float* W2  = (const float*)d_in[7];
    const float* al2 = (const float*)d_in[8];
    const float* ar2 = (const float*)d_in[9];
    const float* b2  = (const float*)d_in[10];
    const float* Wp  = (const float*)d_in[11];
    const float* bp  = (const float*)d_in[12];
    const int* row = ei;
    const int* col = ei + EE;
    float* out = (float*)d_out;

    // carve workspace (256B-aligned regions)
    char* p = (char*)d_ws;
    auto carve = [&](size_t bytes) -> void* {
        void* r = (void*)p;
        p += (bytes + 255) & ~(size_t)255;
        return r;
    };
    _Float16* XH  = (_Float16*)carve((size_t)NN * FF * sizeof(_Float16)); // 12.8MB
    _Float16* WT  = (_Float16*)carve((size_t)FF * FF * sizeof(_Float16)); // 32KB
    float*    WH  = (float*)   carve((size_t)NN * FF * sizeof(float));    // 25.6MB
    float*    EL  = (float*)   carve((size_t)NN * NH * sizeof(float));
    float*    ER  = (float*)   carve((size_t)NN * NH * sizeof(float));
    float*    LW  = (float*)   carve((size_t)EE * sizeof(float));         // 3.2MB
    float*    S   = (float*)   carve((size_t)EE * NH * sizeof(float));    // 12.8MB
    unsigned* ME  = (unsigned*)carve((size_t)NN * NH * sizeof(unsigned));
    float*    DEN = (float*)   carve((size_t)NN * NH * sizeof(float));
    float*    ACC = (float*)   carve((size_t)NN * FF * sizeof(float));    // 25.6MB

    const int T = 256;
    const int gNF  = (NN * FF + T - 1) / T;   // 25000
    const int gWW  = (FF * FF + T - 1) / T;
    const int gNH  = (NN * NH + T - 1) / T;
    const int gE   = (EE + T - 1) / T;
    const int gEH  = (EE * NH + T - 1) / T;
    const int gE32 = (EE * 32 + T - 1) / T;   // 100000
    const int gN   = (NN + T - 1) / T;
    const int gMM  = NN / 16;                 // 3125 (exact)

    // shared pre-pass
    k_logw<<<gE, T, 0, stream>>>(ew, LW);

    // ---------------- layer 1 ----------------
    k_cvt_x<<<gNF, T, 0, stream>>>(x, XH, NN * FF);
    k_cvt_w<<<gWW, T, 0, stream>>>(W1, WT);
    k_gemm <<<gMM, T, 0, stream>>>(XH, WT, WH);
    k_elr  <<<gNH, T, 0, stream>>>(WH, al1, ar1, EL, ER);
    k_init <<<gNF, T, 0, stream>>>(ME, DEN, ACC);
    k_edge1<<<gEH, T, 0, stream>>>(row, col, EL, ER, LW, S, ME);
    k_edge2<<<gEH, T, 0, stream>>>(col, S, ME, DEN);
    k_edge3<<<gE32, T, 0, stream>>>(row, col, S, DEN, WH, ACC);
    k_fin1 <<<gNF, T, 0, stream>>>(ACC, b1, XH);   // relu(h1) -> f16, layer-2 input

    // ---------------- layer 2 ----------------
    k_cvt_w<<<gWW, T, 0, stream>>>(W2, WT);
    k_gemm <<<gMM, T, 0, stream>>>(XH, WT, WH);
    k_elr  <<<gNH, T, 0, stream>>>(WH, al2, ar2, EL, ER);
    k_init <<<gNF, T, 0, stream>>>(ME, DEN, ACC);
    k_edge1<<<gEH, T, 0, stream>>>(row, col, EL, ER, LW, S, ME);
    k_edge2<<<gEH, T, 0, stream>>>(col, S, ME, DEN);
    k_edge3<<<gE32, T, 0, stream>>>(row, col, S, DEN, WH, ACC);

    // ---------------- projection + softmax ----------------
    k_proj <<<gN, T, 0, stream>>>(ACC, b2, Wp, bp, out);
}
